// get_model_27376121545185
// MI455X (gfx1250) — matrix operand order repaired and checked by reading, not compiled, not run on hardware
//
#include <hip/hip_runtime.h>

// ---------------------------------------------------------------------------
// CDNA5 (gfx1250, wave32) implementation of the FMR point-cloud segmentation
// network.  GEMM-shaped compute goes through v_wmma_f32_16x16x32_f16 (f16 in,
// f32 accumulate).  A-tiles of the generic GEMM are moved by the Tensor Data
// Mover (tensor_load_to_lds + s_wait_tensorcnt); B-tiles are staged into LDS
// transposed (float4 global loads on the in-bounds fast path) so fragments
// load as contiguous ds_load_b128.  Geometry math (FPS / kNN / interpolation
// weights) stays f32 VALU.
// ---------------------------------------------------------------------------

typedef _Float16 v16h __attribute__((ext_vector_type(16)));
typedef float    v8f  __attribute__((ext_vector_type(8)));

#define B_   8
#define N_   4096

// ---- Tensor Data Mover (guarded: device pass + builtin present) -----------
#if defined(__HIP_DEVICE_COMPILE__) && defined(__has_builtin)
#if __has_builtin(__builtin_amdgcn_tensor_load_to_lds) && \
    __has_builtin(__builtin_amdgcn_s_wait_tensorcnt)
#define HAVE_TDM 1
#endif
#endif

#ifdef HAVE_TDM
typedef unsigned int v4u __attribute__((ext_vector_type(4)));
typedef int          v8i __attribute__((ext_vector_type(8)));
typedef int          v4i __attribute__((ext_vector_type(4)));

// 2D tile load: tile_dim1 rows x tile_dim0 elems (f32), row stride = stride
// elems; reads past (tdim0, tdim1) return zero (replaces bounds guards).
// This toolchain's builtin takes 6 args (4 SGPR groups + spare group + cpol).
__device__ __forceinline__ void tdm_load_2d_f32(unsigned lds_off, const float* gptr,
                                                unsigned rows, unsigned cols,
                                                unsigned stride, unsigned tdim0,
                                                unsigned tdim1)
{
    unsigned long long ga = (unsigned long long)(size_t)gptr;
    v4u g0;
    g0.x = __builtin_amdgcn_readfirstlane(1u);                    // count=1, user D#
    g0.y = __builtin_amdgcn_readfirstlane(lds_off);               // lds_addr
    g0.z = __builtin_amdgcn_readfirstlane((unsigned)(ga & 0xffffffffu));
    g0.w = __builtin_amdgcn_readfirstlane(
               (unsigned)((ga >> 32) & 0x01ffffffu) | (2u << 30)); // type=2 (image)
    v8i g1;
    g1[0] = __builtin_amdgcn_readfirstlane((int)(2u << 16));      // data_size=4B, mask=0
    g1[1] = __builtin_amdgcn_readfirstlane((int)((tdim0 & 0xffffu) << 16));
    g1[2] = __builtin_amdgcn_readfirstlane((int)((tdim0 >> 16) | ((tdim1 & 0xffffu) << 16)));
    g1[3] = __builtin_amdgcn_readfirstlane((int)((tdim1 >> 16) | ((cols & 0xffffu) << 16)));
    g1[4] = __builtin_amdgcn_readfirstlane((int)(rows & 0xffffu)); // tile_dim1 (tile_dim2=0)
    g1[5] = __builtin_amdgcn_readfirstlane((int)stride);           // tensor_dim0_stride[31:0]
    g1[6] = 0;                                                     // stride[47:32] | dim1_stride
    g1[7] = 0;
    v4i z4 = { 0, 0, 0, 0 };
    v8i z8 = { 0, 0, 0, 0, 0, 0, 0, 0 };
    __builtin_amdgcn_tensor_load_to_lds(g0, g1, z4, z4, z8, 0);
}
#endif

__constant__ float c_fp9[27] = {
     1.f,  1.f,  1.f,   1.f,  1.f, -1.f,   1.f, -1.f,  1.f,
     1.f, -1.f, -1.f,  -1.f,  1.f,  1.f,  -1.f,  1.f, -1.f,
    -1.f, -1.f,  1.f,  -1.f, -1.f, -1.f,   0.f,  0.f,  0.f };

// ------------------------------------------------------------------ transpose
__global__ void k_transpose(const float* __restrict__ xyz, float* __restrict__ p0)
{
    long total = (long)B_ * 3 * N_;
    for (long i = blockIdx.x * (long)blockDim.x + threadIdx.x; i < total;
         i += (long)gridDim.x * blockDim.x) {
        int n = i % N_;  long bc = i / N_;  int c = bc % 3;  int b = bc / 3;
        p0[((long)b * N_ + n) * 3 + c] = xyz[((long)b * 3 + c) * N_ + n];
    }
}

// ------------------------------------------------------------------ FPS
__global__ void k_fps(const float* __restrict__ pts, int* __restrict__ fidx,
                      int Np, int npoint)
{
    __shared__ float dist[4096];
    __shared__ float rv[256];
    __shared__ int   ri[256];
    __shared__ int   sfar;
    int b = blockIdx.x, tid = threadIdx.x;
    const float* base = pts + (long)b * Np * 3;

    for (int n = tid; n < Np; n += 256) dist[n] = 1.0e10f;
    if (tid == 0) sfar = 0;
    __syncthreads();

    for (int i = 0; i < npoint; ++i) {
        int far = sfar;
        if (tid == 0) fidx[(long)b * npoint + i] = far;
        float cx = base[far * 3 + 0], cy = base[far * 3 + 1], cz = base[far * 3 + 2];
        float bv = -1.f; int bi = 0;
        for (int n = tid; n < Np; n += 256) {
            float dx = base[n * 3 + 0] - cx;
            float dy = base[n * 3 + 1] - cy;
            float dz = base[n * 3 + 2] - cz;
            float d  = dx * dx + dy * dy + dz * dz;
            float dn = fminf(dist[n], d);
            dist[n]  = dn;
            if (dn > bv) { bv = dn; bi = n; }
        }
        rv[tid] = bv; ri[tid] = bi;
        __syncthreads();
        for (int s = 128; s > 0; s >>= 1) {
            if (tid < s) {
                if (rv[tid + s] > rv[tid] ||
                    (rv[tid + s] == rv[tid] && ri[tid + s] < ri[tid])) {
                    rv[tid] = rv[tid + s]; ri[tid] = ri[tid + s];
                }
            }
            __syncthreads();
        }
        if (tid == 0) sfar = ri[0];
        __syncthreads();
    }
}

// ------------------------------------------------------------------ gather xyz
__global__ void k_gather3(const float* __restrict__ pts, const int* __restrict__ idx,
                          float* __restrict__ out, int Np, int S)
{
    int total = B_ * S;
    for (int i = blockIdx.x * blockDim.x + threadIdx.x; i < total;
         i += gridDim.x * blockDim.x) {
        int b = i / S;
        int id = idx[i];
        const float* p = pts + ((long)b * Np + id) * 3;
        out[(long)i * 3 + 0] = p[0];
        out[(long)i * 3 + 1] = p[1];
        out[(long)i * 3 + 2] = p[2];
    }
}

// ------------------------------------------------------------------ brute kNN
__global__ void k_knn(const float* __restrict__ q, const float* __restrict__ ref,
                      int* __restrict__ nidx, int S, int Nref, int K)
{
    int total = B_ * S;
    for (int i = blockIdx.x * blockDim.x + threadIdx.x; i < total;
         i += gridDim.x * blockDim.x) {
        int b = i / S;
        float qx = q[(long)i * 3 + 0], qy = q[(long)i * 3 + 1], qz = q[(long)i * 3 + 2];
        float bd[32]; int bi[32];
        for (int k = 0; k < K; ++k) { bd[k] = 3.0e38f; bi[k] = 0; }
        const float* rb = ref + (long)b * Nref * 3;
        for (int n = 0; n < Nref; ++n) {
            float dx = rb[n * 3 + 0] - qx, dy = rb[n * 3 + 1] - qy, dz = rb[n * 3 + 2] - qz;
            float d = dx * dx + dy * dy + dz * dz;
            if (d < bd[K - 1]) {
                int p = K - 1;
                while (p > 0 && bd[p - 1] > d) { bd[p] = bd[p - 1]; bi[p] = bi[p - 1]; --p; }
                bd[p] = d; bi[p] = n;
            }
        }
        for (int k = 0; k < K; ++k) nidx[(long)i * K + k] = bi[k];
    }
}

// ------------------------------------------------------------------ FMR m1 + g
__global__ void k_fmr_m1_g(const float* __restrict__ xyzp, const float* __restrict__ ptsp,
                           const float* __restrict__ nxyz, const int* __restrict__ nidx,
                           const float* __restrict__ W1, const float* __restrict__ b1,
                           const float* __restrict__ W2, const float* __restrict__ b2,
                           float* __restrict__ g, int S, int K, int Nprev, int Cprev,
                           int cin, int h1, float radius, int identity)
{
    __shared__ float sW1[7 * 128], sb1[128], sW2[128], sb2;
    __shared__ float ws[32], srel[32][3];
    __shared__ int   snb[32];

    int bs = blockIdx.x, b = bs / S, lane = threadIdx.x;

    for (int i = lane; i < 7 * h1; i += 32) sW1[i] = W1[i];
    for (int i = lane; i < h1; i += 32) { sb1[i] = b1[i]; sW2[i] = W2[i]; }
    if (lane == 0) sb2 = b2[0];
    __syncthreads();

    float cx = 0.f, cy = 0.f, cz = 0.f;
    if (!identity) {
        cx = nxyz[(long)bs * 3 + 0]; cy = nxyz[(long)bs * 3 + 1]; cz = nxyz[(long)bs * 3 + 2];
    }

    if (lane < K) {
        int nb = identity ? lane : nidx[(long)bs * K + lane];
        snb[lane] = nb;
        const float* pp = xyzp + ((long)b * Nprev + nb) * 3;
        float rx = pp[0] - cx, ry = pp[1] - cy, rz = pp[2] - cz;
        srel[lane][0] = rx; srel[lane][1] = ry; srel[lane][2] = rz;
        float acc = 0.f;
        for (int f = 0; f < 9; ++f) {
            float dx = rx - c_fp9[f * 3 + 0] * radius;
            float dy = ry - c_fp9[f * 3 + 1] * radius;
            float dz = rz - c_fp9[f * 3 + 2] * radius;
            float dist = sqrtf(dx * dx + dy * dy + dz * dz + 1e-12f);
            float enc[7] = { rx, ry, rz, dx, dy, dz, dist };
            float o = sb2;
            for (int j = 0; j < h1; ++j) {
                float h = sb1[j];
                #pragma unroll
                for (int i = 0; i < 7; ++i) h += enc[i] * sW1[i * h1 + j];
                o += fmaxf(h, 0.f) * sW2[j];
            }
            acc += o;
        }
        ws[lane] = acc;
    }
    __syncthreads();

    float inv = 1.0f / (9.0f * (float)K);
    for (int c = lane; c < cin; c += 32) {
        float acc = 0.f;
        for (int k = 0; k < K; ++k) {
            float fv;
            if (cin != Cprev)
                fv = (c < 3) ? srel[k][c]
                             : ptsp[((long)b * Nprev + snb[k]) * Cprev + (c - 3)];
            else
                fv = ptsp[((long)b * Nprev + snb[k]) * Cprev + c];
            acc += ws[k] * fv;
        }
        g[(long)bs * cin + c] = acc * inv;
    }
}

// ------------------------------------------------------------------ FMR m2+max (WMMA)
// one block (4 waves) per (b,s).  H chunk (16 x h2, f16) in LDS; W2 tiles
// staged into LDS transposed (BhT[col][k]) so B fragments are contiguous
// ds_load_b128; channel-max folded via shfl_xor(16).
__global__ void k_fmr_m2max(const float* __restrict__ g, const float* __restrict__ W1,
                            const float* __restrict__ b1, const float* __restrict__ W2,
                            const float* __restrict__ b2, float* __restrict__ out,
                            int cin, int h2, int cout)
{
    __shared__ float    sg[512], sW1[256], sb1[256], omax[1024];
    __shared__ _Float16 Hc[16 * 256];
    __shared__ _Float16 BhT[64 * 32];

    int bs = blockIdx.x, tid = threadIdx.x;
    int wave = tid >> 5, lane = tid & 31, half = lane >> 4, row = lane & 15;

    for (int i = tid; i < cin;  i += 128) sg[i]  = g[(long)bs * cin + i];
    for (int i = tid; i < h2;   i += 128) { sW1[i] = W1[i]; sb1[i] = b1[i]; }
    for (int i = tid; i < cout; i += 128) omax[i] = -3.0e38f;
    __syncthreads();

    int kpad = ((h2 + 31) >> 5) << 5;          // h2 rounded up to 32

    for (int c0 = 0; c0 < cin; c0 += 16) {
        // H chunk: H[r][j] = relu(g[c0+r]*W1[j]+b1[j]); pad rows/cols -> 0
        for (int e = tid; e < 16 * kpad; e += 128) {
            int r = e / kpad, j = e % kpad;
            float hv = 0.f;
            if (j < h2 && c0 + r < cin)
                hv = fmaxf(sg[c0 + r] * sW1[j] + sb1[j], 0.f);
            Hc[r * 256 + j] = (_Float16)hv;
        }
        __syncthreads();

        for (int nb = 0; nb < cout; nb += 64) {
            v8f acc = { 0.f, 0.f, 0.f, 0.f, 0.f, 0.f, 0.f, 0.f };
            for (int k0 = 0; k0 < kpad; k0 += 32) {
                // stage W2[k0..k0+31, nb..nb+63] transposed to f16
                if (k0 + 32 <= h2) {
                    // fast path: whole tile in bounds -> float4 (b128) loads
                    for (int e = tid; e < 512; e += 128) {
                        int k = e >> 4, cq = (e & 15) << 2;
                        float4 vv = *(const float4*)&W2[(size_t)(k0 + k) * cout + nb + cq];
                        BhT[(cq + 0) * 32 + k] = (_Float16)vv.x;
                        BhT[(cq + 1) * 32 + k] = (_Float16)vv.y;
                        BhT[(cq + 2) * 32 + k] = (_Float16)vv.z;
                        BhT[(cq + 3) * 32 + k] = (_Float16)vv.w;
                    }
                } else {
                    for (int e = tid; e < 2048; e += 128) {
                        int k = e >> 6, c = e & 63, kk = k0 + k;
                        BhT[c * 32 + k] =
                            (_Float16)((kk < h2) ? W2[(size_t)kk * cout + nb + c] : 0.f);
                    }
                }
                __syncthreads();

                v16h a, bf;
                #pragma unroll
                for (int u = 0; u < 8; ++u) {
                    a[u]     = Hc[row * 256 + k0 + half * 16 + u];
                    a[8 + u] = Hc[row * 256 + k0 + half * 16 + 8 + u];
                }
                #pragma unroll
                for (int u = 0; u < 16; ++u)
                    bf[u] = BhT[(wave * 16 + row) * 32 + half * 16 + u];

                acc = __builtin_amdgcn_wmma_f32_16x16x32_f16(
                          false, a, false, bf, (short)0, acc, false, false);
                __syncthreads();
            }
            // max over this 16-row chunk (mask padded rows)
            float m = -3.0e38f;
            #pragma unroll
            for (int e = 0; e < 8; ++e) {
                int c = c0 + e + half * 8;
                if (c < cin) m = fmaxf(m, acc[e]);
            }
            m = fmaxf(m, __shfl_xor(m, 16));
            if (half == 0) {
                int col = nb + wave * 16 + row;
                omax[col] = fmaxf(omax[col], m);
            }
        }
        __syncthreads();
    }

    for (int o = tid; o < cout; o += 128)
        out[(long)bs * cout + o] = fmaxf(omax[o] + b2[o], 0.f);
}

// ------------------------------------------------------------------ generic WMMA GEMM
// Y[M,N] = act(X[M,K] @ W[K,N] + bias[N]);  grid (M/16, N/64), block 128.
// A tile (16x32 f32) is DMA'd by the Tensor Data Mover; B tile staged
// transposed as f16 (float4 loads on the in-bounds fast path).
__global__ void k_gemm_bias_act(const float* __restrict__ X, const float* __restrict__ W,
                                const float* __restrict__ bias, float* __restrict__ Y,
                                int M, int K, int N, int dorelu)
{
    __shared__ float    As[16 * 32];
    __shared__ _Float16 BhT[64 * 32];

    int m0 = blockIdx.x * 16, nb = blockIdx.y * 64;
    int tid = threadIdx.x, wave = tid >> 5, lane = tid & 31, half = lane >> 4, row = lane & 15;

    v8f acc = { 0.f, 0.f, 0.f, 0.f, 0.f, 0.f, 0.f, 0.f };

    for (int k0 = 0; k0 < K; k0 += 32) {
        __syncthreads();   // previous tile fully consumed before overwrite
#ifdef HAVE_TDM
        if (wave == 0)     // wave-level DMA: OOB (tdim) reads return zero
            tdm_load_2d_f32((unsigned)(size_t)&As[0], &X[(size_t)m0 * K + k0],
                            16u, 32u, (unsigned)K,
                            (unsigned)(K - k0), (unsigned)(M - m0));
#else
        for (int e = tid; e < 512; e += 128) {
            int r = e >> 5, k = e & 31;
            As[e] = (m0 + r < M && k0 + k < K) ? X[(size_t)(m0 + r) * K + k0 + k] : 0.f;
        }
#endif
        if (k0 + 32 <= K && nb + 64 <= N && (N & 3) == 0) {
            // fast path: whole 32x64 tile in bounds -> float4 (b128) loads
            for (int e = tid; e < 512; e += 128) {
                int k = e >> 4, cq = (e & 15) << 2;
                float4 vv = *(const float4*)&W[(size_t)(k0 + k) * N + nb + cq];
                BhT[(cq + 0) * 32 + k] = (_Float16)vv.x;
                BhT[(cq + 1) * 32 + k] = (_Float16)vv.y;
                BhT[(cq + 2) * 32 + k] = (_Float16)vv.z;
                BhT[(cq + 3) * 32 + k] = (_Float16)vv.w;
            }
        } else {
            for (int e = tid; e < 2048; e += 128) {
                int k = e >> 6, c = e & 63;
                BhT[c * 32 + k] = (_Float16)((k0 + k < K && nb + c < N)
                                                 ? W[(size_t)(k0 + k) * N + nb + c] : 0.f);
            }
        }
        if (k0 + 32 < K && m0 + row < M)
            __builtin_prefetch(&X[(size_t)(m0 + row) * K + k0 + 32], 0, 1);
#ifdef HAVE_TDM
        if (wave == 0) __builtin_amdgcn_s_wait_tensorcnt(0);
#endif
        __syncthreads();

        v16h a, bf;
        #pragma unroll
        for (int u = 0; u < 8; ++u) {
            a[u]     = (_Float16)As[row * 32 + half * 16 + u];
            a[8 + u] = (_Float16)As[row * 32 + half * 16 + 8 + u];
        }
        #pragma unroll
        for (int u = 0; u < 16; ++u)
            bf[u] = BhT[(wave * 16 + row) * 32 + half * 16 + u];

        acc = __builtin_amdgcn_wmma_f32_16x16x32_f16(
                  false, a, false, bf, (short)0, acc, false, false);
    }

    #pragma unroll
    for (int e = 0; e < 8; ++e) {
        int m = m0 + e + half * 8;
        int n = nb + wave * 16 + row;
        if (m < M && n < N) {
            float v = acc[e] + bias[n];
            if (dorelu) v = fmaxf(v, 0.f);
            Y[(size_t)m * N + n] = v;
        }
    }
}

// ------------------------------------------------------------------ 3-NN + weights
__global__ void k_knn3(const float* __restrict__ xyz1, const float* __restrict__ xyz2,
                       int* __restrict__ idx3, float* __restrict__ w3, int N1, int S2)
{
    int total = B_ * N1;
    for (int i = blockIdx.x * blockDim.x + threadIdx.x; i < total;
         i += gridDim.x * blockDim.x) {
        int b = i / N1;
        float qx = xyz1[(long)i * 3 + 0], qy = xyz1[(long)i * 3 + 1], qz = xyz1[(long)i * 3 + 2];
        float d0 = 3e38f, d1 = 3e38f, d2 = 3e38f; int i0 = 0, i1 = 0, i2 = 0;
        const float* rb = xyz2 + (long)b * S2 * 3;
        for (int s = 0; s < S2; ++s) {
            float dx = rb[s * 3 + 0] - qx, dy = rb[s * 3 + 1] - qy, dz = rb[s * 3 + 2] - qz;
            float d = dx * dx + dy * dy + dz * dz;
            if (d < d0)      { d2 = d1; i2 = i1; d1 = d0; i1 = i0; d0 = d; i0 = s; }
            else if (d < d1) { d2 = d1; i2 = i1; d1 = d;  i1 = s; }
            else if (d < d2) { d2 = d;  i2 = s; }
        }
        float w0 = 1.f / fmaxf(d0, 1e-10f);
        float w1 = 1.f / fmaxf(d1, 1e-10f);
        float w2 = 1.f / fmaxf(d2, 1e-10f);
        float ws = w0 + w1 + w2;
        idx3[(long)i * 3 + 0] = i0; idx3[(long)i * 3 + 1] = i1; idx3[(long)i * 3 + 2] = i2;
        w3[(long)i * 3 + 0] = w0 / ws; w3[(long)i * 3 + 1] = w1 / ws; w3[(long)i * 3 + 2] = w2 / ws;
    }
}

// ------------------------------------------------------------------ interp + concat
__global__ void k_interp_concat(const float* __restrict__ pts1, const float* __restrict__ pts2,
                                const int* __restrict__ idx3, const float* __restrict__ w3,
                                float* __restrict__ X, int N1, int C1, int S2, int C2)
{
    int Ct = C1 + C2;
    long total = (long)B_ * N1 * Ct;
    for (long i = blockIdx.x * (long)blockDim.x + threadIdx.x; i < total;
         i += (long)gridDim.x * blockDim.x) {
        int c = i % Ct; long bn = i / Ct; int n = bn % N1; int b = bn / N1;
        float v;
        if (c < C1) v = pts1[((long)b * N1 + n) * C1 + c];
        else {
            int c2 = c - C1; long q = (long)b * N1 + n;
            float acc = 0.f;
            #pragma unroll
            for (int k = 0; k < 3; ++k)
                acc += w3[q * 3 + k] * pts2[((long)b * S2 + idx3[q * 3 + k]) * C2 + c2];
            v = acc;
        }
        X[i] = v;
    }
}

// fp6: pts2 has S==1 -> broadcast
__global__ void k_concat_bcast(const float* __restrict__ pts1, const float* __restrict__ pts2,
                               float* __restrict__ X, int N1, int C1, int C2)
{
    int Ct = C1 + C2;
    long total = (long)B_ * N1 * Ct;
    for (long i = blockIdx.x * (long)blockDim.x + threadIdx.x; i < total;
         i += (long)gridDim.x * blockDim.x) {
        int c = i % Ct; long bn = i / Ct; int n = bn % N1; int b = bn / N1;
        X[i] = (c < C1) ? pts1[((long)b * N1 + n) * C1 + c]
                        : pts2[(long)b * C2 + (c - C1)];
    }
}

// fp1 input: [cls_oh(16) | p0(3) | p0(3) | interp(128)] = 150 channels
__global__ void k_fp1_concat(const float* __restrict__ cls, const float* __restrict__ p0,
                             const float* __restrict__ pts2, const int* __restrict__ idx3,
                             const float* __restrict__ w3, float* __restrict__ X,
                             int S2, int C2)
{
    int Ct = 22 + C2;
    long total = (long)B_ * N_ * Ct;
    for (long i = blockIdx.x * (long)blockDim.x + threadIdx.x; i < total;
         i += (long)gridDim.x * blockDim.x) {
        int c = i % Ct; long bn = i / Ct; int n = bn % N_; int b = bn / N_;
        float v;
        if (c < 16)      v = cls[b * 16 + c];
        else if (c < 19) v = p0[((long)b * N_ + n) * 3 + (c - 16)];
        else if (c < 22) v = p0[((long)b * N_ + n) * 3 + (c - 19)];
        else {
            int c2 = c - 22; long q = (long)b * N_ + n;
            float acc = 0.f;
            #pragma unroll
            for (int k = 0; k < 3; ++k)
                acc += w3[q * 3 + k] * pts2[((long)b * S2 + idx3[q * 3 + k]) * C2 + c2];
            v = acc;
        }
        X[i] = v;
    }
}

// ---------------------------------------------------------------------------
extern "C" void kernel_launch(void* const* d_in, const int* in_sizes, int n_in,
                              void* d_out, int out_size, void* d_ws, size_t ws_size,
                              hipStream_t stream)
{
    const float* xyz = (const float*)d_in[0];
    const float* cls = (const float*)d_in[1];
    const float* P[80];
    for (int i = 2; i < 80 && i < n_in; ++i) P[i] = (const float*)d_in[i];

    char*  ws  = (char*)d_ws;
    size_t off = 0;
    auto alloc = [&](size_t elems) -> float* {
        float* p = (float*)(ws + off);
        off = (off + elems * 4 + 255) & ~(size_t)255;
        return p;
    };
    auto alloci = [&](size_t elems) -> int* { return (int*)alloc(elems); };

    float* outF  = (float*)d_out;
    float* l3ptr = outF + (size_t)B_ * N_ * 50;           // second tuple element

    // ----- p0 = transpose(xyz)
    float* p0 = alloc((size_t)B_ * N_ * 3);
    k_transpose<<<(B_ * 3 * N_ + 255) / 256, 256, 0, stream>>>(xyz, p0);

    struct Lvl { int Nprev, Cprev, cin, h1, h2, cout, S, K; float r; int pw; int ident; };
    Lvl L[6] = {
        { 4096,   3,   6,  16,  16,   64, 1024, 32, 0.15f,  2, 0 },
        { 1024,  64,  64,  32,  64,  128,  512, 32, 0.15f, 10, 0 },
        {  512, 128, 128,  32,  64,  256,  256, 16, 0.25f, 18, 0 },
        {  256, 256, 256,  64, 128,  256,  128, 16, 0.40f, 26, 0 },
        {  128, 256, 256,  64, 128,  512,   32,  8, 0.60f, 34, 0 },
        {   32, 512, 512, 128, 256, 1024,    1, 32, 10.0f, 42, 1 },
    };

    const float* xs[7]; const float* fs[7];
    xs[0] = p0; fs[0] = p0;

    for (int l = 0; l < 6; ++l) {
        Lvl& v = L[l];
        float* nx = nullptr; int* nid = nullptr;
        if (!v.ident) {
            int* fidx = alloci((size_t)B_ * v.S);
            nx  = alloc((size_t)B_ * v.S * 3);
            nid = alloci((size_t)B_ * v.S * v.K);
            k_fps<<<B_, 256, 0, stream>>>(xs[l], fidx, v.Nprev, v.S);
            k_gather3<<<(B_ * v.S + 255) / 256, 256, 0, stream>>>(xs[l], fidx, nx, v.Nprev, v.S);
            k_knn<<<(B_ * v.S + 255) / 256, 256, 0, stream>>>(nx, xs[l], nid, v.S, v.Nprev, v.K);
        }
        float* g    = alloc((size_t)B_ * v.S * v.cin);
        float* fout = alloc((size_t)B_ * v.S * v.cout);
        k_fmr_m1_g<<<B_ * v.S, 32, 0, stream>>>(
            xs[l], fs[l], nx, nid, P[v.pw], P[v.pw + 1], P[v.pw + 2], P[v.pw + 3],
            g, v.S, v.K, v.Nprev, v.Cprev, v.cin, v.h1, v.r, v.ident);
        k_fmr_m2max<<<B_ * v.S, 128, 0, stream>>>(
            g, P[v.pw + 4], P[v.pw + 5], P[v.pw + 6], P[v.pw + 7], fout,
            v.cin, v.h2, v.cout);
        xs[l + 1] = nx;       // unused for level 6 (S==1 branch ignores coords)
        fs[l + 1] = fout;
    }

    auto gemm = [&](const float* X, int M, int K, int N,
                    const float* W, const float* b, float* Y, int relu) {
        dim3 grid((M + 15) / 16, (N + 63) / 64);
        k_gemm_bias_act<<<grid, 128, 0, stream>>>(X, W, b, Y, M, K, N, relu);
    };

    // ----- fp6 : broadcast-concat [f5(512) | f6(1024)] -> 1536 -> 512 -> 512
    {
        int N1 = 32;
        float* X  = alloc((size_t)B_ * N1 * 1536);
        float* t  = alloc((size_t)B_ * N1 * 512);
        float* up = alloc((size_t)B_ * N1 * 512);
        long tot = (long)B_ * N1 * 1536;
        k_concat_bcast<<<(int)((tot + 255) / 256), 256, 0, stream>>>(fs[5], fs[6], X, N1, 512, 1024);
        gemm(X, B_ * N1, 1536, 512, P[50], P[51], t, 1);
        gemm(t, B_ * N1, 512, 512, P[52], P[53], up, 1);
        fs[6] = up;   // "up" after fp6, lives at level-5 resolution
    }
    // ----- fp5 : (B,128) <- (B,32);  [f4(256) | interp(512)] -> 768 -> 256 -> 256
    float* up5;
    {
        int N1 = 128, S2 = 32;
        int*   i3 = alloci((size_t)B_ * N1 * 3);
        float* w3 = alloc((size_t)B_ * N1 * 3);
        float* X  = alloc((size_t)B_ * N1 * 768);
        float* t  = alloc((size_t)B_ * N1 * 256);
        up5       = alloc((size_t)B_ * N1 * 256);
        k_knn3<<<(B_ * N1 + 255) / 256, 256, 0, stream>>>(xs[4], xs[5], i3, w3, N1, S2);
        long tot = (long)B_ * N1 * 768;
        k_interp_concat<<<(int)((tot + 255) / 256), 256, 0, stream>>>(fs[4], fs[6], i3, w3, X, N1, 256, S2, 512);
        gemm(X, B_ * N1, 768, 256, P[54], P[55], t, 1);
        gemm(t, B_ * N1, 256, 256, P[56], P[57], up5, 1);
    }
    // ----- fp4 : (B,256) <- (B,128); [f3(256) | interp(256)] -> 512 -> 256 -> 256 == l3
    {
        int N1 = 256, S2 = 128;
        int*   i3 = alloci((size_t)B_ * N1 * 3);
        float* w3 = alloc((size_t)B_ * N1 * 3);
        float* X  = alloc((size_t)B_ * N1 * 512);
        float* t  = alloc((size_t)B_ * N1 * 256);
        k_knn3<<<(B_ * N1 + 255) / 256, 256, 0, stream>>>(xs[3], xs[4], i3, w3, N1, S2);
        long tot = (long)B_ * N1 * 512;
        k_interp_concat<<<(int)((tot + 255) / 256), 256, 0, stream>>>(fs[3], up5, i3, w3, X, N1, 256, S2, 256);
        gemm(X, B_ * N1, 512, 256, P[58], P[59], t, 1);
        gemm(t, B_ * N1, 256, 256, P[60], P[61], l3ptr, 1);   // l3 straight into d_out
    }
    // ----- fp3 : (B,512) <- (B,256); [f2(128) | interp(256)] -> 384 -> 256 -> 128
    float* up3;
    {
        int N1 = 512, S2 = 256;
        int*   i3 = alloci((size_t)B_ * N1 * 3);
        float* w3 = alloc((size_t)B_ * N1 * 3);
        float* X  = alloc((size_t)B_ * N1 * 384);
        float* t  = alloc((size_t)B_ * N1 * 256);
        up3       = alloc((size_t)B_ * N1 * 128);
        k_knn3<<<(B_ * N1 + 255) / 256, 256, 0, stream>>>(xs[2], xs[3], i3, w3, N1, S2);
        long tot = (long)B_ * N1 * 384;
        k_interp_concat<<<(int)((tot + 255) / 256), 256, 0, stream>>>(fs[2], l3ptr, i3, w3, X, N1, 128, S2, 256);
        gemm(X, B_ * N1, 384, 256, P[62], P[63], t, 1);
        gemm(t, B_ * N1, 256, 128, P[64], P[65], up3, 1);
    }
    // ----- fp2 : (B,1024) <- (B,512); [f1(64) | interp(128)] -> 192 -> 256 -> 128
    float* up2;
    {
        int N1 = 1024, S2 = 512;
        int*   i3 = alloci((size_t)B_ * N1 * 3);
        float* w3 = alloc((size_t)B_ * N1 * 3);
        float* X  = alloc((size_t)B_ * N1 * 192);
        float* t  = alloc((size_t)B_ * N1 * 256);
        up2       = alloc((size_t)B_ * N1 * 128);
        k_knn3<<<(B_ * N1 + 255) / 256, 256, 0, stream>>>(xs[1], xs[2], i3, w3, N1, S2);
        long tot = (long)B_ * N1 * 192;
        k_interp_concat<<<(int)((tot + 255) / 256), 256, 0, stream>>>(fs[1], up3, i3, w3, X, N1, 64, S2, 128);
        gemm(X, B_ * N1, 192, 256, P[66], P[67], t, 1);
        gemm(t, B_ * N1, 256, 128, P[68], P[69], up2, 1);
    }
    // ----- fp1 : (B,4096) <- (B,1024); [cls(16)|p0|p0|interp(128)] -> 150 -> 128 -> 128 -> 128
    float* t1 = alloc((size_t)B_ * N_ * 128);
    float* t2 = alloc((size_t)B_ * N_ * 128);
    {
        int S2 = 1024;
        int*   i3 = alloci((size_t)B_ * N_ * 3);
        float* w3 = alloc((size_t)B_ * N_ * 3);
        float* X  = alloc((size_t)B_ * N_ * 150);
        k_knn3<<<(B_ * N_ + 255) / 256, 256, 0, stream>>>(p0, xs[1], i3, w3, N_, S2);
        long tot = (long)B_ * N_ * 150;
        k_fp1_concat<<<(int)((tot + 255) / 256), 256, 0, stream>>>(cls, p0, up2, i3, w3, X, S2, 128);
        gemm(X,  B_ * N_, 150, 128, P[70], P[71], t1, 1);
        gemm(t1, B_ * N_, 128, 128, P[72], P[73], t2, 1);
        gemm(t2, B_ * N_, 128, 128, P[74], P[75], t1, 1);
    }
    // ----- head : 128 -> 128 (relu) -> 50 (no relu)  -> x_out
    gemm(t1, B_ * N_, 128, 128, P[76], P[77], t2, 1);
    gemm(t2, B_ * N_, 128,  50, P[78], P[79], outF, 0);

    (void)in_sizes; (void)out_size; (void)ws_size;
}
